// MHA_layer_69715909149424
// MI455X (gfx1250) — compile-verified, matrix operand order checked by reading, not run on hardware
//
#include <hip/hip_runtime.h>

typedef __attribute__((ext_vector_type(16))) _Float16 v16h;
typedef __attribute__((ext_vector_type(8)))  float    v8f;
typedef unsigned int u32x4 __attribute__((ext_vector_type(4)));
typedef int          i32x4 __attribute__((ext_vector_type(4)));
typedef int          i32x8 __attribute__((ext_vector_type(8)));

constexpr int BB = 4;
constexpr int SS = 4096;
constexpr int DD = 1024;

#if __has_builtin(__builtin_amdgcn_tensor_load_to_lds)
#define HAVE_TDM 1
#else
#define HAVE_TDM 0
#endif

union AFrag { v16h h; unsigned int u[8]; };

__device__ __forceinline__ unsigned int pack2(_Float16 a, _Float16 b) {
  unsigned short ua = __builtin_bit_cast(unsigned short, a);
  unsigned short ub = __builtin_bit_cast(unsigned short, b);
  return (unsigned int)ua | ((unsigned int)ub << 16);
}

// K index of pair j (0..7) for this lane in the 16x32 f16 A/B fragment layout.
// lanes 0-15: K = {0..7, 16..23}; lanes 16-31: +8.
__device__ __forceinline__ int kpair(int j, int lane) {
  return ((j < 4) ? (2 * j) : (8 + 2 * j)) + ((lane >> 4) << 3);
}

// Load a 16x32 f16 A/B fragment from row-major storage (element (K,row) at
// p[row*ld + coloff + K]).  Works for both LDS and global pointers.
__device__ __forceinline__ v16h ld_frag(const _Float16* __restrict__ p, int row,
                                        int ld, int coloff, int lane) {
  AFrag f;
  const _Float16* r = p + (size_t)row * ld + coloff;
#pragma unroll
  for (int j = 0; j < 8; ++j) {
    f.u[j] = *(const unsigned int*)(r + kpair(j, lane));
  }
  return f.h;
}

__device__ __forceinline__ v8f wmma_f16(v16h a, v16h b, v8f c) {
  return __builtin_amdgcn_wmma_f32_16x16x32_f16(false, a, false, b, (short)0, c,
                                                false, false);
}

// ---------------------------------------------------------------------------
// TDM: async DMA of a 32x1024 f16 row-major tile (row stride 1024 elems) from
// global memory into LDS, with pad_enable inserting 1 DWORD after every 256
// DWORDs stored -> LDS row stride = 2056 B = 1028 halves, mid-row pad of 2
// halves at d==512.  Bank stride between rows = 514 % 64 = 2 -> the 16
// row-strided lanes of a WMMA fragment read hit 16 distinct banks.
// ---------------------------------------------------------------------------
#define LDK_T 1028  // padded LDS row stride in halves
__device__ __forceinline__ int padadj(int d) { return d + ((d >> 9) << 1); }

#if HAVE_TDM
__device__ __forceinline__ void tdm_load_tile_32xD(unsigned int lds_byte_off,
                                                   const _Float16* gsrc) {
  unsigned long long ga = (unsigned long long)(size_t)gsrc;
  u32x4 g0;
  g0[0] = 1u;                                     // count=1 (valid user D#)
  g0[1] = lds_byte_off;                           // lds_addr
  g0[2] = (unsigned int)ga;                       // global_addr[31:0]
  g0[3] = (unsigned int)(ga >> 32) | (2u << 30);  // global_addr[56:32]|type=2
  i32x8 g1;
  g1[0] = 0x01D10000;  // data_size=2B, pad_enable, pad_interval=256dw, pad=1dw
  g1[1] = 0x04000000;  // tensor_dim0 = 1024
  g1[2] = 0x00200000;  // tensor_dim1 = 32
  g1[3] = 0x04000000;  // tile_dim0 = 1024
  g1[4] = 32;          // tile_dim1 = 32
  g1[5] = 1024;        // tensor_dim0_stride = 1024
  g1[6] = 0;
  g1[7] = 0;
  i32x4 z4 = {0, 0, 0, 0};
#if defined(__clang_major__) && (__clang_major__ >= 23)
  i32x8 z8 = {0, 0, 0, 0, 0, 0, 0, 0};
  __builtin_amdgcn_tensor_load_to_lds(g0, g1, z4, z4, z8, 0);
#else
  __builtin_amdgcn_tensor_load_to_lds(g0, g1, z4, z4, 0);
#endif
}
#endif

// ---------------------------------------------------------------------------
// Kernel 1: fused QKV projection + bias + rotary, fp32 in -> f16 out.
// grid = (DD/128, (BB*SS)/64, 3), block = 256 (8 waves).
// ---------------------------------------------------------------------------
#define LDX 34  // padded row stride (halves) for 32-wide K tiles

__global__ __launch_bounds__(256) void qkv_rope_kernel(
    const float* __restrict__ x,
    const float* __restrict__ wq, const float* __restrict__ wk,
    const float* __restrict__ wv,
    const float* __restrict__ bq, const float* __restrict__ bk,
    const float* __restrict__ bv,
    const float* __restrict__ cosb, const float* __restrict__ sinb,
    _Float16* __restrict__ Qh, _Float16* __restrict__ Kh,
    _Float16* __restrict__ Vh) {
  __shared__ _Float16 xa[64 * LDX];
  __shared__ _Float16 wb[128 * LDX];

  const int z = blockIdx.z;
  const float* w = (z == 0) ? wq : (z == 1) ? wk : wv;
  const float* bias = (z == 0) ? bq : (z == 1) ? bk : bv;
  _Float16* out = (z == 0) ? Qh : (z == 1) ? Kh : Vh;
  const int do_rope = (z < 2);

  const int tid = threadIdx.x;
  const int lane = tid & 31;
  const int wave = tid >> 5;
  const int wm = wave & 3;
  const int wn = wave >> 2;
  const int m0 = blockIdx.y * 64;
  const int n0 = blockIdx.x * 128;

  v8f acc[4] = {};

  for (int k0 = 0; k0 < DD; k0 += 32) {
    for (int i = tid; i < 1024; i += 256) {
      int r = i >> 4;
      int c = (i & 15) * 2;
      float2 v = *(const float2*)(x + (size_t)(m0 + r) * DD + k0 + c);
      *(unsigned int*)(xa + r * LDX + c) = pack2((_Float16)v.x, (_Float16)v.y);
    }
    for (int i = tid; i < 2048; i += 256) {
      int r = i >> 4;
      int c = (i & 15) * 2;
      float2 v = *(const float2*)(w + (size_t)(n0 + r) * DD + k0 + c);
      *(unsigned int*)(wb + r * LDX + c) = pack2((_Float16)v.x, (_Float16)v.y);
    }
    __syncthreads();

    v16h a = ld_frag(xa, wm * 16 + (lane & 15), LDX, 0, lane);
#pragma unroll
    for (int t = 0; t < 4; ++t) {
      v16h b = ld_frag(wb, wn * 64 + t * 16 + (lane & 15), LDX, 0, lane);
      acc[t] = wmma_f16(a, b, acc[t]);
    }
    __syncthreads();
  }

  const int ncol = lane & 15;
  const int rbase = (lane >> 4) * 8;
#pragma unroll
  for (int t = 0; t < 4; ++t) {
    const int n_g = n0 + wn * 64 + t * 16 + ncol;
    const float bval = bias[n_g];
#pragma unroll
    for (int r = 0; r < 8; ++r) {
      const int row_g = m0 + wm * 16 + rbase + r;
      float v = acc[t][r] + bval;
      if (do_rope) {
        const int spos = row_g & (SS - 1);
        const int fi = n_g >> 1;
        const float c = cosb[(size_t)spos * (DD / 2) + fi];
        const float s = sinb[(size_t)spos * (DD / 2) + fi];
        const float p = __shfl_xor(v, 1, 32);
        v = (n_g & 1) ? (p * s + v * c) : (v * c - p * s);
      }
      out[(size_t)row_g * DD + n_g] = (_Float16)v;
    }
  }
}

// ---------------------------------------------------------------------------
// Kernel 2: fused attention  out = (Q K^T / B) V.
// grid = (SS/32, BB), block = 256 (8 waves).  32 query rows x D=1024 out cols.
// K/V tiles DMA'd to LDS by the Tensor Data Mover, double-issued and gated
// with s_wait_tensorcnt so the V DMA overlaps QK^T and the next K DMA
// overlaps S.V.
// ---------------------------------------------------------------------------
__global__ __launch_bounds__(256) void attn_kernel(
    const _Float16* __restrict__ Qh, const _Float16* __restrict__ Kh,
    const _Float16* __restrict__ Vh, float* __restrict__ out) {
  __shared__ alignas(16) _Float16 kvK[32 * LDK_T];  // 64.25 KB, TDM-padded
  __shared__ alignas(16) _Float16 kvV[32 * LDK_T];  // 64.25 KB, TDM-padded
  __shared__ float sS[2][16][33];                   // 32x32 score tile

  const int tid = threadIdx.x;
  const int lane = tid & 31;
  const int wave = tid >> 5;
  const int wr = wave >> 2;  // query row group (16 rows)
  const int wc = wave & 3;   // 256-wide d slice
  const int b = blockIdx.y;
  const int s0 = blockIdx.x * 32;
  const size_t rowbase = (size_t)b * SS + s0;
  const int mrow = (lane >> 4) * 8;

  const _Float16* Kbase = Kh + (size_t)b * SS * DD;
  const _Float16* Vbase = Vh + (size_t)b * SS * DD;

  // Resident Q fragments: rows [wr*16,+16), cols [wc*256,+256)
  v16h qa[8];
  {
    const _Float16* qbase =
        Qh + (rowbase + wr * 16 + (lane & 15)) * (size_t)DD + wc * 256;
#pragma unroll
    for (int kf = 0; kf < 8; ++kf) qa[kf] = ld_frag(qbase, 0, DD, kf * 32, lane);
  }

  v8f acc[16] = {};

#if HAVE_TDM
  const unsigned int kvK_off = (unsigned int)(size_t)(&kvK[0]);
  const unsigned int kvV_off = (unsigned int)(size_t)(&kvV[0]);
  if (wave == 0) {
    tdm_load_tile_32xD(kvK_off, Kbase);  // K(0)
    tdm_load_tile_32xD(kvV_off, Vbase);  // V(0)
  }
#endif

  for (int t0 = 0; t0 < SS; t0 += 32) {
    // ---- phase A: zero score tile; make K(t0) visible
    for (int i = tid; i < 2 * 16 * 33; i += 256) ((float*)sS)[i] = 0.0f;
#if HAVE_TDM
    if (wave == 0) __builtin_amdgcn_s_wait_tensorcnt(1);  // K(t0) done
#else
    for (int i = tid; i < 8192; i += 256) {  // manual staging fallback
      int r = i >> 8;
      int c = (i & 255) * 4;
      *(uint2*)(kvK + r * LDK_T + padadj(c)) =
          *(const uint2*)(Kbase + (size_t)(t0 + r) * DD + c);
    }
#endif
    __syncthreads();

    // ---- phase B: partial Q.K^T over this wave's 256-wide d slice
    v8f sacc[2] = {};
#pragma unroll
    for (int kf = 0; kf < 8; ++kf) {
      const int coladj = padadj(wc * 256 + kf * 32);
#pragma unroll
      for (int sub = 0; sub < 2; ++sub) {
        v16h kb = ld_frag(kvK, sub * 16 + (lane & 15), LDK_T, coladj, lane);
        sacc[sub] = wmma_f16(qa[kf], kb, sacc[sub]);
      }
    }
#pragma unroll
    for (int sub = 0; sub < 2; ++sub)
#pragma unroll
      for (int r = 0; r < 8; ++r)
        atomicAdd(&sS[wr][mrow + r][sub * 16 + (lane & 15)], sacc[sub][r]);
    __syncthreads();

    // ---- phase C: prefetch K(t0+32) via TDM; fold score tile to f16 A-frag
#if HAVE_TDM
    if (wave == 0 && t0 + 32 < SS)
      tdm_load_tile_32xD(kvK_off, Kbase + (size_t)(t0 + 32) * DD);
#else
    for (int i = tid; i < 8192; i += 256) {
      int r = i >> 8;
      int c = (i & 255) * 4;
      *(uint2*)(kvV + r * LDK_T + padadj(c)) =
          *(const uint2*)(Vbase + (size_t)(t0 + r) * DD + c);
    }
#endif
    AFrag sf;
    {
      const int m = lane & 15;
#pragma unroll
      for (int j = 0; j < 8; ++j) {
        int k = kpair(j, lane);
        _Float16 h0 = (_Float16)(sS[wr][m][k] * 0.25f);  // /B scale
        _Float16 h1 = (_Float16)(sS[wr][m][k + 1] * 0.25f);
        sf.u[j] = pack2(h0, h1);
      }
    }
#if HAVE_TDM
    if (wave == 0) {
      if (t0 + 32 < SS)
        __builtin_amdgcn_s_wait_tensorcnt(1);  // V(t0) done, K(t0+32) in flight
      else
        __builtin_amdgcn_s_wait_tensorcnt(0);  // last block: V(t0) done
    }
#endif
    __syncthreads();

    // ---- phase D: acc += S_tile (16x32) . V (32 x 256)
#pragma unroll
    for (int n = 0; n < 16; ++n) {
      const int dadj = padadj(wc * 256 + n * 16 + (lane & 15));
      AFrag vf;
#pragma unroll
      for (int j = 0; j < 8; ++j) {
        int k = kpair(j, lane);  // K dim = t (row), pairs stride LDK_T
        unsigned short lo =
            __builtin_bit_cast(unsigned short, kvV[(size_t)k * LDK_T + dadj]);
        unsigned short hi = __builtin_bit_cast(
            unsigned short, kvV[(size_t)(k + 1) * LDK_T + dadj]);
        vf.u[j] = (unsigned int)lo | ((unsigned int)hi << 16);
      }
      acc[n] = wmma_f16(sf.h, vf.h, acc[n]);
    }
    __syncthreads();

#if HAVE_TDM
    if (wave == 0 && t0 + 32 < SS)
      tdm_load_tile_32xD(kvV_off, Vbase + (size_t)(t0 + 32) * DD);
#endif
  }

  // write out fp32
#pragma unroll
  for (int n = 0; n < 16; ++n) {
    const int dg = wc * 256 + n * 16 + (lane & 15);
#pragma unroll
    for (int r = 0; r < 8; ++r) {
      const int row = s0 + wr * 16 + mrow + r;
      out[((size_t)b * SS + row) * DD + dg] = acc[n][r];
    }
  }
}

extern "C" void kernel_launch(void* const* d_in, const int* in_sizes, int n_in,
                              void* d_out, int out_size, void* d_ws,
                              size_t ws_size, hipStream_t stream) {
  (void)in_sizes; (void)n_in; (void)out_size; (void)ws_size;
  const float* x  = (const float*)d_in[0];
  const float* fc = (const float*)d_in[1];
  const float* fs = (const float*)d_in[2];
  const float* wq = (const float*)d_in[3];
  const float* bq = (const float*)d_in[4];
  const float* wk = (const float*)d_in[5];
  const float* bk = (const float*)d_in[6];
  const float* wv = (const float*)d_in[7];
  const float* bv = (const float*)d_in[8];
  float* out = (float*)d_out;

  _Float16* Qh = (_Float16*)d_ws;
  _Float16* Kh = Qh + (size_t)BB * SS * DD;
  _Float16* Vh = Kh + (size_t)BB * SS * DD;

  dim3 g1(DD / 128, (BB * SS) / 64, 3);
  qkv_rope_kernel<<<g1, dim3(256), 0, stream>>>(x, wq, wk, wv, bq, bk, bv, fc,
                                                fs, Qh, Kh, Vh);
  dim3 g2(SS / 32, BB);
  attn_kernel<<<g2, dim3(256), 0, stream>>>(Qh, Kh, Vh, out);
}
